// TrajGRUCell_83159156785908
// MI455X (gfx1250) — matrix operand order repaired and checked by reading, not compiled, not run on hardware
//
#include <hip/hip_runtime.h>
#include <hip/hip_bf16.h>

typedef __bf16 bf16;
typedef __attribute__((ext_vector_type(16))) __bf16 v16bf;
typedef __attribute__((ext_vector_type(8)))  float  v8f;

#if defined(__has_builtin)
#if __has_builtin(__builtin_amdgcn_tensor_load_to_lds) && __has_builtin(__builtin_amdgcn_s_wait_tensorcnt)
#define HAVE_TDM 1
#endif
#endif
#ifndef HAVE_TDM
#define HAVE_TDM 0
#endif

#if HAVE_TDM
typedef __attribute__((ext_vector_type(4))) unsigned int tdm_u32x4;
typedef __attribute__((ext_vector_type(8))) int          tdm_i32x8;
typedef __attribute__((ext_vector_type(4))) int          tdm_i32x4;
#endif

#define B_    8
#define CIN   8
#define HID   64
#define LNK   9
#define HH    120
#define WW    120
#define NPIX  (HH*WW)        // 14400
#define NTOT  (B_*NPIX)      // 115200
#define OCH   (3*HID)        // 192
#define KH2H  (LNK*HID)      // 576
#define KI2H  96             // 72 padded to 96 (3 WMMA K-steps)

// ---------------------------------------------------------------------------
// Generic bf16 WMMA GEMM: C[M][N] = A[M][K] * B[K][N], fp32 accumulate.
// Block tile 64x128x32, 256 threads = 8 waves, each wave 16(M)x64(N).
// A tile staged by the Tensor Data Mover (row-major + HW padding -> stride 40
// halves, bank-conflict free); B tile staged K-major (transposed) manually so
// each fragment is two ds_load_b128.
// ---------------------------------------------------------------------------
__global__ __launch_bounds__(256)
void gemm_bf16_wmma(const bf16* __restrict__ A, const bf16* __restrict__ Bm,
                    float* __restrict__ C, int M, int N, int K)
{
    __shared__ bf16 sA [64 * 40];    // row-major, padded stride 40 halves
    __shared__ bf16 sBT[128 * 40];   // K-major (transposed), padded stride 40

    const int tid  = threadIdx.x;
    const int lane = tid & 31;
    const int wave = tid >> 5;
    const int bm = blockIdx.y * 64;
    const int bn = blockIdx.x * 128;
    const int wm = (wave >> 1) * 16;   // 0,16,32,48
    const int wn = (wave & 1) * 64;    // 0,64

    const v8f vz = {0.f,0.f,0.f,0.f,0.f,0.f,0.f,0.f};
    v8f acc[4] = {vz, vz, vz, vz};

    // B staging assignment
    const int br = tid >> 3;           // B row (K) 0..31
    const int bc = (tid & 7) * 16;     // B col 0..112
    const bf16* gB = Bm + (size_t)br * N + bn + bc;

#if HAVE_TDM
    // ---- TDM descriptor for the A tile: 2D tensor [M][K] bf16, tile 32(K)x64(M),
    // LDS padding: 4 DWORDs after every 16 DWORDs (row = 64B) -> stride 40 halves.
    const unsigned ldsA = (unsigned)(uintptr_t)(&sA[0]);
    tdm_i32x8 g1;
    g1[0] = (1 << 16)          // data_size = 2 bytes
          | (1 << 20)          // pad_enable
          | (3 << 22)          // pad_interval: 16 DWORDs
          | (3 << 25);         // pad_amount : 4 DWORDs
    g1[1] = (int)((unsigned)(K & 0xffff) << 16);                       // tensor_dim0[15:0]
    g1[2] = (int)((((unsigned)K >> 16) & 0xffffu) |
                  ((unsigned)(M & 0xffff) << 16));                     // dim0[31:16] | dim1[15:0]
    g1[3] = (int)(((((unsigned)M) >> 16) & 0xffffu) | (32u << 16));    // dim1[31:16] | tile_dim0=32
    g1[4] = 64;                                                        // tile_dim1=64, tile_dim2=0
    g1[5] = K;                                                         // tensor_dim0_stride[31:0]
    g1[6] = 0;
    g1[7] = 0;
    const tdm_i32x4 gz4 = {0, 0, 0, 0};
    const tdm_i32x8 gz8 = {0, 0, 0, 0, 0, 0, 0, 0};
#else
    const int ar = tid >> 2;           // A row 0..63
    const int ac = (tid & 3) * 8;      // A col 0,8,16,24
    const bf16* gA = A + (size_t)(bm + ar) * K + ac;
#endif

    for (int k0 = 0; k0 < K; k0 += 32) {
        uint4 b4l = *(const uint4*)(gB + (size_t)k0 * N);
        uint4 b4h = *(const uint4*)(gB + (size_t)k0 * N + 8);
#if !HAVE_TDM
        uint4 a4  = *(const uint4*)(gA + k0);
#endif
        if (k0 + 32 < K)
            __builtin_prefetch(gB + (size_t)(k0 + 32) * N, 0, 1);      // global_prefetch_b8

        __syncthreads();   // prior iteration's fragment reads complete

#if HAVE_TDM
        if (wave == 0) {   // one TDM op per block per K-step (EXEC ignored by TDM)
            unsigned long long ga = (unsigned long long)(uintptr_t)A +
                                    ((unsigned long long)bm * (unsigned)K + (unsigned)k0) * 2ull;
            tdm_u32x4 g0;
            g0[0] = 1u;                                   // count=1, user descriptor
            g0[1] = ldsA;                                 // LDS destination
            g0[2] = (unsigned)(ga & 0xffffffffu);         // global_addr[31:0]
            g0[3] = (unsigned)((ga >> 32) & 0x01ffffffu) | (2u << 30);  // addr[56:32] | type=2
            __builtin_amdgcn_tensor_load_to_lds(g0, g1, gz4, gz4, gz8, 0);
        }
#else
        *(uint4*)(&sA[ar * 40 + ac]) = a4;
#endif
        // scatter B into K-major LDS
        {
            const bf16* bl = (const bf16*)&b4l;
            const bf16* bh = (const bf16*)&b4h;
            #pragma unroll
            for (int i = 0; i < 8; ++i) sBT[(bc + i)     * 40 + br] = bl[i];
            #pragma unroll
            for (int i = 0; i < 8; ++i) sBT[(bc + 8 + i) * 40 + br] = bh[i];
        }
#if HAVE_TDM
        if (wave == 0) __builtin_amdgcn_s_wait_tensorcnt(0);
#endif
        __syncthreads();

        // A fragment: lane = row (M), lane[4] selects K 0-15/16-31 halves,
        // vector elem e carries K 16*lane[4] + e (matches B fragment order).
        v16bf af;
        {
            const bf16* p = &sA[(wm + (lane & 15)) * 40 + ((lane >> 4) << 4)];
            #pragma unroll
            for (int e = 0; e < 8; ++e) af[e]     = p[e];
            #pragma unroll
            for (int e = 0; e < 8; ++e) af[8 + e] = p[8 + e];
        }
        // Batch all four B fragments, then issue WMMAs back-to-back so the
        // scheduler can use partial DS waits + XDL co-execution.
        v16bf bfb[4];
        #pragma unroll
        for (int j = 0; j < 4; ++j) {
            const bf16* p = &sBT[(wn + j * 16 + (lane & 15)) * 40 + ((lane >> 4) << 4)];
            #pragma unroll
            for (int e = 0; e < 16; ++e) bfb[j][e] = p[e];
        }
        #pragma unroll
        for (int j = 0; j < 4; ++j)
            acc[j] = __builtin_amdgcn_wmma_f32_16x16x32_bf16(
                false, af, false, bfb[j], (short)0, acc[j], false, false);
    }

    // C/D layout: VGPR r -> M = r + 8*lane[4], N = lane[3:0]
    const int crow0 = bm + wm + ((lane >> 4) << 3);
    const int ccol0 = bn + wn + (lane & 15);
    #pragma unroll
    for (int j = 0; j < 4; ++j)
        #pragma unroll
        for (int r = 0; r < 8; ++r)
            C[(size_t)(crow0 + r) * N + ccol0 + j * 16] = acc[j][r];
}

// ---------------------------------------------------------------------------
// Weight preparation (fp32 -> bf16, i2h repacked to GEMM A [192][96], K pad)
// ---------------------------------------------------------------------------
__global__ void cast_h2h_w_kernel(const float* __restrict__ w, bf16* __restrict__ o)
{
    int i = blockIdx.x * 256 + threadIdx.x;   // 192*576
    o[i] = (bf16)w[i];
}

__global__ void pack_i2h_w_kernel(const float* __restrict__ w, bf16* __restrict__ o)
{
    int i = blockIdx.x * 256 + threadIdx.x;   // 192*96
    int oc = i / KI2H, k = i % KI2H;
    float v = (k < 72) ? w[oc * 72 + k] : 0.f;   // k = c*9 + ky*3 + kx
    o[i] = (bf16)v;
}

// ---------------------------------------------------------------------------
// im2col of x (3x3, pad 1) into bf16 [96][115200]
// ---------------------------------------------------------------------------
__global__ void im2col_x_kernel(const float* __restrict__ x, bf16* __restrict__ Xc)
{
    int n = blockIdx.x * 256 + threadIdx.x;   // 0..115199
    int k = blockIdx.y;                       // 0..95
    float v = 0.f;
    if (k < 72) {
        int c = k / 9, t = k % 9;
        int dy = t / 3 - 1, dx = t % 3 - 1;
        int b = n / NPIX, p = n % NPIX;
        int y = p / WW + dy, xx = p % WW + dx;
        if (y >= 0 && y < HH && xx >= 0 && xx < WW)
            v = x[(size_t)(b * CIN + c) * NPIX + y * WW + xx];
    }
    Xc[(size_t)k * NTOT + n] = (bf16)v;
}

// ---------------------------------------------------------------------------
// Flow generator stage 1: f = relu(conv5x5(x;i2f) + conv5x5(h_prev;h2f))
// 16x16 pixel tile per block, channel-chunked 20x20 LDS patches.
// ---------------------------------------------------------------------------
__global__ __launch_bounds__(256)
void flowgen_f_kernel(const float* __restrict__ x, const float* __restrict__ hprev,
                      const float* __restrict__ i2f_w, const float* __restrict__ i2f_b,
                      const float* __restrict__ h2f_w, const float* __restrict__ h2f_b,
                      float* __restrict__ f)
{
    __shared__ float sP[6400];                 // 20*20*16 = 25.6 KB
    const int b = blockIdx.y;
    const int ty = blockIdx.x >> 3, tx = blockIdx.x & 7;
    const int tid = threadIdx.x;
    const int lx = tid & 15, ly = tid >> 4;
    const int gx = tx * 16 + lx, gy = ty * 16 + ly;

    float acc[32];
    #pragma unroll
    for (int oc = 0; oc < 32; ++oc) acc[oc] = 0.f;

    // ---- x phase (8 channels) ----
    for (int i = tid; i < 3200; i += 256) {
        int c = i / 400, pp = i % 400;
        int py = pp / 20, px = pp % 20;
        int sy = ty * 16 + py - 2, sx = tx * 16 + px - 2;
        float v = 0.f;
        if (sx >= 0 && sx < WW && sy >= 0 && sy < HH)
            v = x[(size_t)(b * CIN + c) * NPIX + sy * WW + sx];
        sP[i] = v;
    }
    __syncthreads();
    for (int c = 0; c < CIN; ++c)
        for (int ky = 0; ky < 5; ++ky)
            for (int kx = 0; kx < 5; ++kx) {
                float v = sP[c * 400 + (ly + ky) * 20 + (lx + kx)];
                #pragma unroll
                for (int oc = 0; oc < 32; ++oc)
                    acc[oc] = fmaf(v, i2f_w[(oc * CIN + c) * 25 + ky * 5 + kx], acc[oc]);
            }

    // ---- h_prev phase (4 chunks of 16 channels) ----
    for (int cc = 0; cc < 4; ++cc) {
        __syncthreads();
        for (int i = tid; i < 6400; i += 256) {
            int c = i / 400, pp = i % 400;
            int py = pp / 20, px = pp % 20;
            int sy = ty * 16 + py - 2, sx = tx * 16 + px - 2;
            float v = 0.f;
            if (sx >= 0 && sx < WW && sy >= 0 && sy < HH)
                v = hprev[(size_t)(b * HID + cc * 16 + c) * NPIX + sy * WW + sx];
            sP[i] = v;
        }
        __syncthreads();
        for (int c = 0; c < 16; ++c)
            for (int ky = 0; ky < 5; ++ky)
                for (int kx = 0; kx < 5; ++kx) {
                    float v = sP[c * 400 + (ly + ky) * 20 + (lx + kx)];
                    int ch = cc * 16 + c;
                    #pragma unroll
                    for (int oc = 0; oc < 32; ++oc)
                        acc[oc] = fmaf(v, h2f_w[(oc * HID + ch) * 25 + ky * 5 + kx], acc[oc]);
                }
    }

    if (gx < WW && gy < HH) {
        #pragma unroll
        for (int oc = 0; oc < 32; ++oc) {
            float r = acc[oc] + i2f_b[oc] + h2f_b[oc];
            f[(size_t)(b * 32 + oc) * NPIX + gy * WW + gx] = fmaxf(r, 0.f);
        }
    }
}

// ---------------------------------------------------------------------------
// Flow generator stage 2: flows = conv5x5(f; flows_w) -> [B][18][H][W]
// ---------------------------------------------------------------------------
__global__ __launch_bounds__(256)
void flows_conv_kernel(const float* __restrict__ f,
                       const float* __restrict__ fw, const float* __restrict__ fb,
                       float* __restrict__ flows)
{
    __shared__ float sP[6400];
    const int b = blockIdx.y;
    const int ty = blockIdx.x >> 3, tx = blockIdx.x & 7;
    const int tid = threadIdx.x;
    const int lx = tid & 15, ly = tid >> 4;
    const int gx = tx * 16 + lx, gy = ty * 16 + ly;

    float acc[18];
    #pragma unroll
    for (int oc = 0; oc < 18; ++oc) acc[oc] = 0.f;

    for (int cc = 0; cc < 2; ++cc) {
        __syncthreads();
        for (int i = tid; i < 6400; i += 256) {
            int c = i / 400, pp = i % 400;
            int py = pp / 20, px = pp % 20;
            int sy = ty * 16 + py - 2, sx = tx * 16 + px - 2;
            float v = 0.f;
            if (sx >= 0 && sx < WW && sy >= 0 && sy < HH)
                v = f[(size_t)(b * 32 + cc * 16 + c) * NPIX + sy * WW + sx];
            sP[i] = v;
        }
        __syncthreads();
        for (int c = 0; c < 16; ++c)
            for (int ky = 0; ky < 5; ++ky)
                for (int kx = 0; kx < 5; ++kx) {
                    float v = sP[c * 400 + (ly + ky) * 20 + (lx + kx)];
                    int ch = cc * 16 + c;
                    #pragma unroll
                    for (int oc = 0; oc < 18; ++oc)
                        acc[oc] = fmaf(v, fw[(oc * 32 + ch) * 25 + ky * 5 + kx], acc[oc]);
                }
    }

    if (gx < WW && gy < HH) {
        #pragma unroll
        for (int oc = 0; oc < 18; ++oc)
            flows[(size_t)(b * 18 + oc) * NPIX + gy * WW + gx] = acc[oc] + fb[oc];
    }
}

// ---------------------------------------------------------------------------
// Bilinear warp of h_prev by flow l, writing bf16 GEMM activations
// Acol[(l*64+c)][b*14400+p]  (coalesced column writes).
// ---------------------------------------------------------------------------
__global__ __launch_bounds__(256)
void warp_kernel(const float* __restrict__ hprev, const float* __restrict__ flows,
                 bf16* __restrict__ Acol)
{
    const int n = blockIdx.x * 256 + threadIdx.x;   // 0..115199
    const int l = blockIdx.y;                       // 0..8
    const int b = n / NPIX, p = n % NPIX;
    const int y = p / WW, x = p % WW;

    const float fx = flows[(size_t)(b * 18 + 2 * l)     * NPIX + p];
    const float fy = flows[(size_t)(b * 18 + 2 * l + 1) * NPIX + p];
    const float sx = (float)x + fx, sy = (float)y + fy;
    const float x0f = floorf(sx), y0f = floorf(sy);
    const int   x0 = (int)x0f, y0 = (int)y0f;
    const float wx1 = sx - x0f, wx0 = 1.f - wx1;
    const float wy1 = sy - y0f, wy0 = 1.f - wy1;

    const int xc0 = min(max(x0, 0), WW - 1),     xc1 = min(max(x0 + 1, 0), WW - 1);
    const int yc0 = min(max(y0, 0), HH - 1),     yc1 = min(max(y0 + 1, 0), HH - 1);
    const float vx0 = (x0 >= 0 && x0 < WW) ? 1.f : 0.f;
    const float vx1 = (x0 + 1 >= 0 && x0 + 1 < WW) ? 1.f : 0.f;
    const float vy0 = (y0 >= 0 && y0 < HH) ? 1.f : 0.f;
    const float vy1 = (y0 + 1 >= 0 && y0 + 1 < HH) ? 1.f : 0.f;
    const float w00 = wy0 * wx0 * vy0 * vx0, w01 = wy0 * wx1 * vy0 * vx1;
    const float w10 = wy1 * wx0 * vy1 * vx0, w11 = wy1 * wx1 * vy1 * vx1;
    const int i00 = yc0 * WW + xc0, i01 = yc0 * WW + xc1;
    const int i10 = yc1 * WW + xc0, i11 = yc1 * WW + xc1;

    const float* hp = hprev + (size_t)b * HID * NPIX;
    for (int c = 0; c < HID; ++c) {
        const float* pc = hp + (size_t)c * NPIX;
        float v = w00 * pc[i00] + w01 * pc[i01] + w10 * pc[i10] + w11 * pc[i11];
        Acol[(size_t)(l * HID + c) * NTOT + n] = (bf16)v;
    }
}

// ---------------------------------------------------------------------------
// GRU gate fusion, float4-vectorized (memory bound, 176 MB read @ HBM rate):
// h_new = u*h_prev + (1-u)*tanh(im + r*hm)
// ---------------------------------------------------------------------------
__global__ __launch_bounds__(256)
void gates_kernel(const float* __restrict__ i2h, const float* __restrict__ h2h,
                  const float* __restrict__ i2h_b, const float* __restrict__ h2h_b,
                  const float* __restrict__ hprev, float* __restrict__ out)
{
    const int t = blockIdx.x * 256 + threadIdx.x;     // < 8*64*14400/4
    const int idx4 = t * 4;
    const int b = idx4 / (HID * NPIX);
    const int r = idx4 % (HID * NPIX);
    const int c = r / NPIX, p = r % NPIX;             // p multiple of 4
    const size_t col = (size_t)b * NPIX + p;

    const float4 ir4 = *(const float4*)(i2h + (size_t)(c)          * NTOT + col);
    const float4 iu4 = *(const float4*)(i2h + (size_t)(HID + c)    * NTOT + col);
    const float4 im4 = *(const float4*)(i2h + (size_t)(2*HID + c)  * NTOT + col);
    const float4 hr4 = *(const float4*)(h2h + (size_t)(c)          * NTOT + col);
    const float4 hu4 = *(const float4*)(h2h + (size_t)(HID + c)    * NTOT + col);
    const float4 hm4 = *(const float4*)(h2h + (size_t)(2*HID + c)  * NTOT + col);
    const float4 hp4 = *(const float4*)(hprev + idx4);
    const float br_ = i2h_b[c] + h2h_b[c];
    const float bu_ = i2h_b[HID + c] + h2h_b[HID + c];
    const float bm_ = i2h_b[2 * HID + c] + h2h_b[2 * HID + c];

    float4 o4;
    {
        const float* irp = (const float*)&ir4; const float* iup = (const float*)&iu4;
        const float* imp = (const float*)&im4; const float* hrp = (const float*)&hr4;
        const float* hup = (const float*)&hu4; const float* hmp = (const float*)&hm4;
        const float* hpp = (const float*)&hp4; float* op = (float*)&o4;
        #pragma unroll
        for (int i = 0; i < 4; ++i) {
            const float rg = 1.f / (1.f + __expf(-(irp[i] + hrp[i] + br_)));
            const float ug = 1.f / (1.f + __expf(-(iup[i] + hup[i] + bu_)));
            const float nm = tanhf(imp[i] + bm_ + rg * hmp[i]);
            op[i] = ug * hpp[i] + (1.f - ug) * nm;
        }
    }
    *(float4*)(out + idx4) = o4;
}

// ---------------------------------------------------------------------------
extern "C" void kernel_launch(void* const* d_in, const int* in_sizes, int n_in,
                              void* d_out, int out_size, void* d_ws, size_t ws_size,
                              hipStream_t stream)
{
    const float* x       = (const float*)d_in[0];
    const float* h_prev  = (const float*)d_in[1];
    const float* i2h_w   = (const float*)d_in[2];
    const float* i2h_b   = (const float*)d_in[3];
    const float* h2h_w   = (const float*)d_in[4];
    const float* h2h_b   = (const float*)d_in[5];
    const float* i2f_w   = (const float*)d_in[6];
    const float* i2f_b   = (const float*)d_in[7];
    const float* h2f_w   = (const float*)d_in[8];
    const float* h2f_b   = (const float*)d_in[9];
    const float* flows_w = (const float*)d_in[10];
    const float* flows_b = (const float*)d_in[11];
    float* out = (float*)d_out;

    // workspace partition (all sizes are 256B multiples)
    char* ws = (char*)d_ws;
    float* f_buf    = (float*)ws;  ws += (size_t)B_ * 32 * NPIX * 4;        // 14.75 MB
    float* flows_b_ = (float*)ws;  ws += (size_t)B_ * 18 * NPIX * 4;        //  8.29 MB
    bf16*  Acol     = (bf16*) ws;  ws += (size_t)KH2H * NTOT * 2;           // 132.7 MB
    bf16*  Xcol     = (bf16*) ws;  ws += (size_t)KI2H * NTOT * 2;           // 22.1 MB
    bf16*  Wh       = (bf16*) ws;  ws += (size_t)OCH * KH2H * 2;            // 221 KB
    bf16*  Wi       = (bf16*) ws;  ws += (size_t)OCH * KI2H * 2;            // 37 KB
    float* i2h_out  = (float*)ws;  ws += (size_t)OCH * NTOT * 4;            // 88.5 MB
    float* h2h_out  = (float*)ws;  ws += (size_t)OCH * NTOT * 4;            // 88.5 MB

    // weight prep
    cast_h2h_w_kernel<<<(OCH * KH2H) / 256, 256, 0, stream>>>(h2h_w, Wh);
    pack_i2h_w_kernel<<<(OCH * KI2H) / 256, 256, 0, stream>>>(i2h_w, Wi);

    // im2col of x for the i2h GEMM
    im2col_x_kernel<<<dim3(NTOT / 256, KI2H), 256, 0, stream>>>(x, Xcol);

    // flow generator
    flowgen_f_kernel<<<dim3(64, B_), 256, 0, stream>>>(
        x, h_prev, i2f_w, i2f_b, h2f_w, h2f_b, f_buf);
    flows_conv_kernel<<<dim3(64, B_), 256, 0, stream>>>(
        f_buf, flows_w, flows_b, flows_b_);

    // 9 bilinear warps -> bf16 activation matrix [576][115200]
    warp_kernel<<<dim3(NTOT / 256, LNK), 256, 0, stream>>>(h_prev, flows_b_, Acol);

    // i2h: [192][96] x [96][115200]      (3 K-steps of WMMA)
    gemm_bf16_wmma<<<dim3(NTOT / 128, OCH / 64), 256, 0, stream>>>(
        Wi, Xcol, i2h_out, OCH, NTOT, KI2H);

    // h2h: [192][576] x [576][115200]    (18 K-steps of WMMA, 25.5 GFLOP)
    gemm_bf16_wmma<<<dim3(NTOT / 128, OCH / 64), 256, 0, stream>>>(
        Wh, Acol, h2h_out, OCH, NTOT, KH2H);

    // fused GRU gates
    gates_kernel<<<(B_ * HID * NPIX) / 1024, 256, 0, stream>>>(
        i2h_out, h2h_out, i2h_b, h2h_b, h_prev, out);
}
